// GeometryLoss_10565619548451
// MI455X (gfx1250) — matrix operand order repaired and checked by reading, not compiled, not run on hardware
//
#include <hip/hip_runtime.h>
#include <hip/hip_bf16.h>
#include <math.h>

#define HH 256
#define WW 256
#define PLANE 65536
#define PAIRS 32
#define NINST 16

#define USE_ASYNC_LDS 1

typedef __attribute__((ext_vector_type(16))) _Float16 v16h;
typedef __attribute__((ext_vector_type(8)))  _Float16 v8h;
typedef __attribute__((ext_vector_type(8)))  float    v8f;

// ---------------------------------------------------------------------------
// Per-instance mask statistics: count, sum(x), sum(y)  -> stats[pair*4 + {0,1,2}]
// ---------------------------------------------------------------------------
__global__ void stats_kernel(const int* __restrict__ labels, float* __restrict__ stats) {
    int b = blockIdx.y;
    int y = blockIdx.x;
    int x = threadIdx.x;
    __shared__ float sc[NINST], sx[NINST], sy[NINST];
    if (x < NINST) { sc[x] = 0.f; sx[x] = 0.f; sy[x] = 0.f; }
    __syncthreads();
    int uid = labels[(b * HH + y) * WW + x];
    if (uid > 0) {
        atomicAdd(&sc[uid - 1], 1.f);
        atomicAdd(&sx[uid - 1], (float)x);
        atomicAdd(&sy[uid - 1], (float)y);
    }
    __syncthreads();
    if (x < NINST && sc[x] != 0.f) {
        atomicAdd(&stats[(b * NINST + x) * 4 + 0], sc[x]);
        atomicAdd(&stats[(b * NINST + x) * 4 + 1], sx[x]);
        atomicAdd(&stats[(b * NINST + x) * 4 + 2], sy[x]);
    }
}

// ---------------------------------------------------------------------------
// Build banded Toeplitz matrix for a 1-D Gaussian (or derivative) correlation
// with symmetric padding folded into the boundary rows.  M32 must be zeroed.
// ---------------------------------------------------------------------------
__global__ void build_toeplitz(float* __restrict__ M32, float sigma, int order, int radius) {
    int i = blockIdx.x;          // output index
    int t = threadIdx.x;         // tap index
    int taps = 2 * radius + 1;
    if (t >= taps) return;
    float s = 0.f;
    for (int u = 0; u < taps; ++u) {
        float xu = (float)(u - radius);
        s += expf(-0.5f * (xu / sigma) * (xu / sigma));
    }
    float xt = (float)(t - radius);
    float w = expf(-0.5f * (xt / sigma) * (xt / sigma)) / s;
    if (order == 1) w *= xt / (sigma * sigma);
    int j = i + (t - radius);
    if (j < 0)   j = -j - 1;          // symmetric padding
    if (j >= WW) j = 2 * WW - j - 1;
    atomicAdd(&M32[i * WW + j], w);
}

__global__ void cvt_mat(const float* __restrict__ M32, _Float16* __restrict__ M16,
                        _Float16* __restrict__ MT16) {
    int idx = blockIdx.x * 256 + threadIdx.x;   // 65536
    int i = idx >> 8, j = idx & 255;
    float v = M32[idx];
    M16[idx] = (_Float16)v;
    MT16[j * WW + i] = (_Float16)v;
}

// ---------------------------------------------------------------------------
// EDT brute force (min-plus), matching the reference exactly.
// ---------------------------------------------------------------------------
__global__ void edt_pass1(const int* __restrict__ labels, float* __restrict__ G) {
    int pair = blockIdx.y; int b = pair >> 4; int uid = (pair & 15) + 1;
    int j = blockIdx.x;         // row
    int x = threadIdx.x;        // col
    __shared__ float f[WW];
    const int* lrow = labels + (b * HH + j) * WW;
    f[x] = (lrow[x] == uid) ? 1e10f : 0.f;
    __syncthreads();
    float best = 3.0e38f;
    for (int xp = 0; xp < WW; ++xp) {
        float d = (float)(x - xp);
        best = fminf(best, f[xp] + d * d);
    }
    G[(pair * HH + j) * WW + x] = best;
}

__global__ void edt_pass2(const int* __restrict__ labels, const float* __restrict__ G,
                          float* __restrict__ DT, unsigned* __restrict__ dtmax) {
    int pair = blockIdx.y; int b = pair >> 4; int uid = (pair & 15) + 1;
    int y = blockIdx.x;
    int x = threadIdx.x;
    const float* Gp = G + pair * PLANE;
    float best = 3.0e38f;
    for (int j = 0; j < HH; ++j) {
        float d = (float)(y - j);
        best = fminf(best, d * d + Gp[j * WW + x]);
    }
    float mf = (labels[(b * HH + y) * WW + x] == uid) ? 1.f : 0.f;
    float dt = sqrtf(fmaxf(best, 0.f)) * mf;
    DT[pair * PLANE + y * WW + x] = dt;
    __shared__ float red[WW];
    red[x] = dt;
    __syncthreads();
    for (int s = 128; s > 0; s >>= 1) {
        if (x < s) red[x] = fmaxf(red[x], red[x + s]);
        __syncthreads();
    }
    if (x == 0) atomicMax(&dtmax[pair], __float_as_uint(red[0]));  // dt >= 0 => bit order ok
}

// ---------------------------------------------------------------------------
// LDS-staged, register-blocked WMMA GEMM:
//   C[pair] (256x256 f32) = A[pair] (f16) x B[pair] (f16), row-major.
//   Pair stride 0 => operand shared across pairs (Toeplitz matrices).
// 4 waves/WG; WG computes a 64x64 tile; wave computes 16x64 (4 accumulators).
// K staged 64-deep per barrier round: 8 back-to-back v_wmma per iteration.
// A tile copied with global_load_async_to_lds_b128 (ASYNCcnt path);
// B tile stored transposed (N-major) so every fragment is a ds_load_b128.
// Staging copies are fully unrolled with constant trip count -> no EXEC
// masking / divergent branches around the async loads.
// ---------------------------------------------------------------------------
__global__ __launch_bounds__(128) void gemm256_wmma(
        const _Float16* __restrict__ A, long long strideA,
        const _Float16* __restrict__ B, long long strideB,
        float* __restrict__ C, long long strideC) {
    const int N = 256;
    int pair = blockIdx.z;
    int tm = blockIdx.y;                  // 64-row tile index (0..3)
    int tn = blockIdx.x;                  // 64-col tile index (0..3)
    const _Float16* Ap = A + (long long)pair * strideA;
    const _Float16* Bp = B + (long long)pair * strideB;
    float* Cp = C + (long long)pair * strideC;

    int t      = threadIdx.x;             // 0..127
    int w      = t >> 5;                  // wave id 0..3
    int lane   = t & 31;
    int half16 = lane >> 4;               // 0: lanes 0-15, 1: lanes 16-31
    int l16    = lane & 15;

    __shared__ alignas(16) _Float16 At[64 * 64];   // M-major, K contiguous (8KB)
    __shared__ alignas(16) _Float16 Bt[64 * 64];   // N-major, K contiguous (8KB)

    v8f acc[4] = {v8f{}, v8f{}, v8f{}, v8f{}};

    for (int kb = 0; kb < N; kb += 64) {
        // ---- stage A tile (64 rows x 64 K halves = 8KB): 4 x 16B chunk/lane ----
#pragma unroll
        for (int u = 0; u < 4; ++u) {
            int c   = t + u * 128;        // 0..511, constant trip count
            int row = c >> 3;             // 0..63
            int off = (c & 7) * 8;        // half offset within K chunk
            const _Float16* g = Ap + (tm * 64 + row) * N + kb + off;
#if USE_ASYNC_LDS
            unsigned ldsa = (unsigned)(unsigned long long)(const void*)&At[row * 64 + off];
            asm volatile("global_load_async_to_lds_b128 %0, %1, off"
                         :: "v"(ldsa), "v"(g) : "memory");
#else
            *(v8h*)&At[row * 64 + off] = *(const v8h*)g;
#endif
        }
        // ---- stage B tile transposed: read 16B along N, scatter ds_store_b16 ----
#pragma unroll
        for (int u = 0; u < 4; ++u) {
            int c  = t + u * 128;         // 0..511
            int k  = c >> 3;              // 0..63
            int n0 = (c & 7) * 8;         // 0..56
            v8h gv = *(const v8h*)(Bp + (kb + k) * N + tn * 64 + n0);
#pragma unroll
            for (int i = 0; i < 8; ++i)
                Bt[(n0 + i) * 64 + k] = gv[i];
        }
#if USE_ASYNC_LDS
        asm volatile("s_wait_asynccnt 0" ::: "memory");
#endif
        __syncthreads();

        // ---- fragments from LDS (all ds_load_b128); 2 K-sub-chunks of 32 ----
        const v8h* Av = (const v8h*)At;
        const v8h* Bv = (const v8h*)Bt;
        int m = w * 16 + l16;
#pragma unroll
        for (int q = 0; q < 2; ++q) {
            v8h alo = Av[m * 8 + q * 4 + half16];       // K 0..7  (lanes 0-15) / 8..15
            v8h ahi = Av[m * 8 + q * 4 + 2 + half16];   // K 16..23 / 24..31
            v16h a;
#pragma unroll
            for (int i = 0; i < 8; ++i) { a[i] = alo[i]; a[8 + i] = ahi[i]; }
#pragma unroll
            for (int j = 0; j < 4; ++j) {
                int n = j * 16 + l16;
                v8h blo = Bv[n * 8 + q * 4 + half16];      // K 0..7  / 8..15
                v8h bhi = Bv[n * 8 + q * 4 + 2 + half16];  // K 16..23 / 24..31
                v16h bfrag;
#pragma unroll
                for (int i = 0; i < 8; ++i) { bfrag[i] = blo[i]; bfrag[8 + i] = bhi[i]; }
                acc[j] = __builtin_amdgcn_wmma_f32_16x16x32_f16(
                    /*neg_a=*/false, a, /*neg_b=*/false, bfrag,
                    /*c_mod=*/(short)0, acc[j], /*reuse_a=*/false, /*reuse_b=*/false);
            }
        }
        __syncthreads();
    }
    // ---- store C: VGPR r -> M = r (lanes 0-15) or r+8 (lanes 16-31); N = lane%16
#pragma unroll
    for (int j = 0; j < 4; ++j) {
#pragma unroll
        for (int r = 0; r < 8; ++r) {
            int m = tm * 64 + w * 16 + r + 8 * half16;
            int n = tn * 64 + j * 16 + l16;
            Cp[m * N + n] = acc[j][r];
        }
    }
}

// ---------------------------------------------------------------------------
// Elementwise helpers
// ---------------------------------------------------------------------------
__global__ void mask_to_f16(const int* __restrict__ labels, _Float16* __restrict__ out) {
    int idx = blockIdx.x * 256 + threadIdx.x;      // PAIRS*PLANE
    int pair = idx >> 16; int pix = idx & 65535;
    int b = pair >> 4; int uid = (pair & 15) + 1;
    out[idx] = (_Float16)((labels[b * PLANE + pix] == uid) ? 1.f : 0.f);
}

__global__ void f32_to_f16(const float* __restrict__ in, _Float16* __restrict__ out, int n) {
    int idx = blockIdx.x * 256 + threadIdx.x;
    if (idx < n) out[idx] = (_Float16)in[idx];
}

__global__ void prod_masked_f16(const float* __restrict__ A, const float* __restrict__ B,
                                const int* __restrict__ labels, _Float16* __restrict__ out) {
    int idx = blockIdx.x * 256 + threadIdx.x;      // PAIRS*PLANE
    int pair = idx >> 16; int pix = idx & 65535;
    int b = pair >> 4; int uid = (pair & 15) + 1;
    float m = (labels[b * PLANE + pix] == uid) ? 1.f : 0.f;
    out[idx] = (_Float16)(A[idx] * B[idx] * m);
}

// ---------------------------------------------------------------------------
// Final structure-tensor combine (masks are disjoint per batch => plain store)
// ---------------------------------------------------------------------------
__global__ void combine_cov(const int* __restrict__ labels, const float* __restrict__ DT,
                            const unsigned* __restrict__ dtmax, const float* __restrict__ NORM,
                            const float* __restrict__ SXX, const float* __restrict__ SXY,
                            const float* __restrict__ SYY, const float* __restrict__ stats,
                            float* __restrict__ covT) {
    int pair = blockIdx.y; int b = pair >> 4; int uid = (pair & 15) + 1;
    int y = blockIdx.x;
    int x = threadIdx.x;
    int pix = y * WW + x;
    if (labels[b * PLANE + pix] != uid) return;
    float cnt   = stats[pair * 4 + 0];
    float valid = (cnt >= 2.f) ? 1.f : 0.f;
    int o = pair * PLANE + pix;
    float nrm = fmaxf(NORM[o], 1e-10f);
    float sxx = SXX[o] / nrm, sxy = SXY[o] / nrm, syy = SYY[o] / nrm;
    float dt   = DT[o];
    float emax = __uint_as_float(dtmax[pair]);
    float w = 0.f;
    if (emax > 1e-6f) { float r = dt / fmaxf(emax, 1e-6f); w = r * r; }
    float iso = 0.5f * (sxx + syy);
    float oxx = ((1.f - w) * sxx + w * iso) * valid;
    float oyy = ((1.f - w) * syy + w * iso) * valid;
    float oxy = ((1.f - w) * sxy) * valid;
    float* cb = covT + (long long)b * 4 * PLANE;
    cb[0 * PLANE + pix] = oxx;
    cb[1 * PLANE + pix] = oxy;
    cb[2 * PLANE + pix] = oxy;
    cb[3 * PLANE + pix] = oyy;
}

// ---------------------------------------------------------------------------
// Direction target (normalized centroid field)
// ---------------------------------------------------------------------------
__global__ void dir_target(const int* __restrict__ labels, const float* __restrict__ stats,
                           float* __restrict__ dirT) {
    int idx = blockIdx.x * 256 + threadIdx.x;      // 2*PLANE
    int b = idx >> 16; int pix = idx & 65535;
    int y = pix >> 8, x = pix & 255;
    int uid = labels[idx];
    float d0 = 0.f, d1 = 0.f;
    if (uid > 0) {
        const float* s = stats + (b * NINST + uid - 1) * 4;
        float c1 = fmaxf(s[0], 1.f);
        float cx = s[1] / c1, cy = s[2] / c1;
        float dx = cx - (float)x, dy = cy - (float)y;
        float n = fmaxf(sqrtf(dx * dx + dy * dy), 1e-6f);
        d0 = dx / n; d1 = dy / n;
    }
    dirT[(long long)b * 2 * PLANE + 0 * PLANE + pix] = d0;
    dirT[(long long)b * 2 * PLANE + 1 * PLANE + pix] = d1;
}

// ---------------------------------------------------------------------------
// Fused loss reduction over all 10 geometry channels
// ---------------------------------------------------------------------------
__global__ void loss_reduce(const float* __restrict__ geom, const float* __restrict__ raw,
                            const float* __restrict__ dirT, const float* __restrict__ covT,
                            float* __restrict__ acc) {
    int idx = blockIdx.x * 256 + threadIdx.x;      // 2*10*PLANE
    int b   = idx / (10 * PLANE);
    int rem = idx % (10 * PLANE);
    int ch  = rem / PLANE;
    int pix = rem % PLANE;
    float g = geom[idx];
    float vdir = 0.f, vcov = 0.f, vraw = 0.f;
    if (ch < 2) {
        float d = fabsf(g - dirT[(b * 2 + ch) * PLANE + pix]);
        vdir = (d < 1.f) ? 0.5f * d * d : d - 0.5f;      // BETA = 1
    } else if (ch < 6) {
        float e = g - covT[(b * 4 + (ch - 2)) * PLANE + pix];
        vcov = e * e;
    } else {
        vraw = fabsf(g - raw[(b * 4 + (ch - 6)) * PLANE + pix]);
    }
    __shared__ float r0[256], r1[256], r2[256];
    int tt = threadIdx.x;
    r0[tt] = vdir; r1[tt] = vcov; r2[tt] = vraw;
    __syncthreads();
    for (int s = 128; s > 0; s >>= 1) {
        if (tt < s) { r0[tt] += r0[tt + s]; r1[tt] += r1[tt + s]; r2[tt] += r2[tt + s]; }
        __syncthreads();
    }
    if (tt == 0) {
        atomicAdd(&acc[0], r0[0]);
        atomicAdd(&acc[1], r1[0]);
        atomicAdd(&acc[2], r2[0]);
    }
}

__global__ void finalize(const float* __restrict__ acc, float* __restrict__ out) {
    out[0] = acc[0] / (float)(2 * 2 * PLANE)
           + acc[1] / (float)(2 * 4 * PLANE)
           + acc[2] / (float)(2 * 4 * PLANE);
}

// ---------------------------------------------------------------------------
// Launch
// ---------------------------------------------------------------------------
extern "C" void kernel_launch(void* const* d_in, const int* in_sizes, int n_in,
                              void* d_out, int out_size, void* d_ws, size_t ws_size,
                              hipStream_t stream) {
    (void)in_sizes; (void)n_in; (void)out_size; (void)ws_size;
    const float* geom   = (const float*)d_in[0];   // (2,10,256,256)
    const float* raw    = (const float*)d_in[1];   // (2,4,256,256)
    const int*   labels = (const int*)d_in[2];     // (2,256,256)
    float* out = (float*)d_out;

    char* p = (char*)d_ws;
    auto alloc = [&](size_t bytes) -> char* {
        char* r = p;
        p += (bytes + 255) & ~(size_t)255;
        return r;
    };
    const size_t PLF32 = (size_t)PAIRS * PLANE * sizeof(float);
    const size_t PLF16 = (size_t)PAIRS * PLANE * sizeof(_Float16);

    float*     stats = (float*)alloc(PAIRS * 4 * sizeof(float));
    unsigned*  dtmax = (unsigned*)alloc(PAIRS * sizeof(unsigned));
    float*     lacc  = (float*)alloc(4 * sizeof(float));
    float*     M32   = (float*)alloc(PLANE * sizeof(float));
    _Float16*  ME    = (_Float16*)alloc(PLANE * 2);
    _Float16*  MET   = (_Float16*)alloc(PLANE * 2);
    _Float16*  MDs   = (_Float16*)alloc(PLANE * 2);
    _Float16*  MDsT  = (_Float16*)alloc(PLANE * 2);
    _Float16*  MDd   = (_Float16*)alloc(PLANE * 2);
    _Float16*  MDdT  = (_Float16*)alloc(PLANE * 2);
    float*     G     = (float*)alloc(PLF32);       // EDT temp, reused as GEMM temp
    float*     TMP   = G;
    float*     DT    = (float*)alloc(PLF32);
    float*     NORM  = (float*)alloc(PLF32);
    float*     GX    = (float*)alloc(PLF32);
    float*     GY    = (float*)alloc(PLF32);
    float*     SXX   = (float*)alloc(PLF32);
    float*     SXY   = (float*)alloc(PLF32);
    float*     SYY   = (float*)alloc(PLF32);
    _Float16*  SA    = (_Float16*)alloc(PLF16);
    _Float16*  SB    = (_Float16*)alloc(PLF16);
    float*     DIRT  = (float*)alloc((size_t)2 * 2 * PLANE * sizeof(float));
    float*     COVT  = (float*)alloc((size_t)2 * 4 * PLANE * sizeof(float));

    const int NPP = PAIRS * PLANE;           // 2 M elements
    const dim3 gPP(NPP / 256), bPP(256);
    const dim3 gRow(HH, PAIRS), bRow(WW);
    const dim3 gGemm(4, 4, PAIRS), bGemm(128);

    hipMemsetAsync(stats, 0, PAIRS * 4 * sizeof(float), stream);
    hipMemsetAsync(dtmax, 0, PAIRS * sizeof(unsigned), stream);
    hipMemsetAsync(lacc,  0, 4 * sizeof(float), stream);
    hipMemsetAsync(COVT,  0, (size_t)2 * 4 * PLANE * sizeof(float), stream);

    // instance statistics
    stats_kernel<<<dim3(HH, 2), WW, 0, stream>>>(labels, stats);

    // Gaussian Toeplitz matrices: sigma=5 r=20 (order 0); sigma_d=5/3 r=7 (orders 0,1)
    const float SIGMA = 5.0f, SIGMA_D = 5.0f / 3.0f;
    hipMemsetAsync(M32, 0, PLANE * sizeof(float), stream);
    build_toeplitz<<<WW, 64, 0, stream>>>(M32, SIGMA, 0, 20);
    cvt_mat<<<PLANE / 256, 256, 0, stream>>>(M32, ME, MET);
    hipMemsetAsync(M32, 0, PLANE * sizeof(float), stream);
    build_toeplitz<<<WW, 64, 0, stream>>>(M32, SIGMA_D, 0, 7);
    cvt_mat<<<PLANE / 256, 256, 0, stream>>>(M32, MDs, MDsT);
    hipMemsetAsync(M32, 0, PLANE * sizeof(float), stream);
    build_toeplitz<<<WW, 64, 0, stream>>>(M32, SIGMA_D, 1, 7);
    cvt_mat<<<PLANE / 256, 256, 0, stream>>>(M32, MDd, MDdT);

    // EDT (min-plus, 2 passes) + per-pair max
    edt_pass1<<<gRow, bRow, 0, stream>>>(labels, G);
    edt_pass2<<<gRow, bRow, 0, stream>>>(labels, G, DT, dtmax);

    // norm = gauss2d(mf, sigma)
    mask_to_f16<<<gPP, bPP, 0, stream>>>(labels, SA);
    gemm256_wmma<<<gGemm, bGemm, 0, stream>>>(ME, 0, SA, PLANE, TMP, PLANE);
    f32_to_f16<<<gPP, bPP, 0, stream>>>(TMP, SB, NPP);
    gemm256_wmma<<<gGemm, bGemm, 0, stream>>>(SB, PLANE, MET, 0, NORM, PLANE);

    // gx = gauss(dt, sigma_d, (0,1)); gy = gauss(dt, sigma_d, (1,0))
    f32_to_f16<<<gPP, bPP, 0, stream>>>(DT, SA, NPP);
    gemm256_wmma<<<gGemm, bGemm, 0, stream>>>(MDs, 0, SA, PLANE, TMP, PLANE);
    f32_to_f16<<<gPP, bPP, 0, stream>>>(TMP, SB, NPP);
    gemm256_wmma<<<gGemm, bGemm, 0, stream>>>(SB, PLANE, MDdT, 0, GX, PLANE);
    gemm256_wmma<<<gGemm, bGemm, 0, stream>>>(MDd, 0, SA, PLANE, TMP, PLANE);
    f32_to_f16<<<gPP, bPP, 0, stream>>>(TMP, SB, NPP);
    gemm256_wmma<<<gGemm, bGemm, 0, stream>>>(SB, PLANE, MDsT, 0, GY, PLANE);

    // sxx / sxy / syy = gauss2d((g*g)*mf, sigma)
    prod_masked_f16<<<gPP, bPP, 0, stream>>>(GX, GX, labels, SA);
    gemm256_wmma<<<gGemm, bGemm, 0, stream>>>(ME, 0, SA, PLANE, TMP, PLANE);
    f32_to_f16<<<gPP, bPP, 0, stream>>>(TMP, SB, NPP);
    gemm256_wmma<<<gGemm, bGemm, 0, stream>>>(SB, PLANE, MET, 0, SXX, PLANE);

    prod_masked_f16<<<gPP, bPP, 0, stream>>>(GX, GY, labels, SA);
    gemm256_wmma<<<gGemm, bGemm, 0, stream>>>(ME, 0, SA, PLANE, TMP, PLANE);
    f32_to_f16<<<gPP, bPP, 0, stream>>>(TMP, SB, NPP);
    gemm256_wmma<<<gGemm, bGemm, 0, stream>>>(SB, PLANE, MET, 0, SXY, PLANE);

    prod_masked_f16<<<gPP, bPP, 0, stream>>>(GY, GY, labels, SA);
    gemm256_wmma<<<gGemm, bGemm, 0, stream>>>(ME, 0, SA, PLANE, TMP, PLANE);
    f32_to_f16<<<gPP, bPP, 0, stream>>>(TMP, SB, NPP);
    gemm256_wmma<<<gGemm, bGemm, 0, stream>>>(SB, PLANE, MET, 0, SYY, PLANE);

    // targets + loss
    combine_cov<<<gRow, bRow, 0, stream>>>(labels, DT, dtmax, NORM, SXX, SXY, SYY, stats, COVT);
    dir_target<<<(2 * PLANE) / 256, 256, 0, stream>>>(labels, stats, DIRT);
    loss_reduce<<<(2 * 10 * PLANE) / 256, 256, 0, stream>>>(geom, raw, DIRT, COVT, lacc);
    finalize<<<1, 1, 0, stream>>>(lacc, out);
}
